// SingleHeadAttention_52020643889270
// MI455X (gfx1250) — matrix operand order repaired and checked by reading, not compiled, not run on hardware
//
#include <hip/hip_runtime.h>
#include <hip/hip_bf16.h>

// ---------------------------------------------------------------------------
// Single-head causal attention, MI455X (gfx1250, wave32, WMMA f16->f32).
//   B=4, S=4096, D_IN=1024, D_HEAD=64
//   q = X@Wq * 1/8 (scale folded into Q), k = X@Wk, v = X@Wv
//   out = softmax(causal(q k^T)) v
// ---------------------------------------------------------------------------

#define BATCH   4
#define SEQ     4096
#define DIN     1024
#define DH      64

typedef _Float16 h16v __attribute__((ext_vector_type(16)));
typedef _Float16 h8v  __attribute__((ext_vector_type(8)));
typedef float    f8v  __attribute__((ext_vector_type(8)));
typedef float    f4v  __attribute__((ext_vector_type(4)));

// D = A*B + C,  16x16x32 f16 -> f32
__device__ __forceinline__ f8v wmma_f16(h16v a, h16v b, f8v c) {
  return __builtin_amdgcn_wmma_f32_16x16x32_f16(
      /*neg_a=*/false, a, /*neg_b=*/false, b,
      /*c_mod=*/(short)0, c, /*reuse_a=*/false, /*reuse_b=*/false);
}

// A-operand (16x32, MxK) gather from a row-major f16 row pointer.
// lane<16: row=lane, K = {0..7, 16..23}; lane>=16: row=lane-16, K = {8..15, 24..31}
__device__ __forceinline__ h16v load_a_f16(const _Float16* rowp, int lane) {
  const int base = (lane < 16) ? 0 : 8;
  h8v lo = *(const h8v*)(rowp + base);
  h8v hi = *(const h8v*)(rowp + base + 16);
  h16v r;
#pragma unroll
  for (int i = 0; i < 8; ++i) { r[i] = lo[i]; r[i + 8] = hi[i]; }
  return r;
}

// Same A gather but from f32 memory (converted to f16 in-register).
__device__ __forceinline__ h16v load_a_f32(const float* rowp, int lane) {
  const int base = (lane < 16) ? 0 : 8;
  f4v x0 = *(const f4v*)(rowp + base);
  f4v x1 = *(const f4v*)(rowp + base + 4);
  f4v x2 = *(const f4v*)(rowp + base + 16);
  f4v x3 = *(const f4v*)(rowp + base + 20);
  h16v r;
#pragma unroll
  for (int i = 0; i < 4; ++i) {
    r[i]      = (_Float16)x0[i];
    r[i + 4]  = (_Float16)x1[i];
    r[i + 8]  = (_Float16)x2[i];
    r[i + 12] = (_Float16)x3[i];
  }
  return r;
}

// B-operand (32x16, KxN): colp points at K=0 of this lane's column (contiguous
// over K).  lane<16 takes K {0..7, 16..23}, lane>=16 takes K {8..15, 24..31}.
__device__ __forceinline__ h16v load_b_f16(const _Float16* colp, int lane) {
  const _Float16* p = colp + ((lane < 16) ? 0 : 8);
  h8v lo = *(const h8v*)(p);
  h8v hi = *(const h8v*)(p + 16);
  h16v r;
#pragma unroll
  for (int i = 0; i < 8; ++i) { r[i] = lo[i]; r[i + 8] = hi[i]; }
  return r;
}

// ---------------------------------------------------------------------------
// Kernel 1: transpose + f16-convert the three [DIN, DH] weights into
//           wt[3][DH][DIN]  (column of output dim becomes contiguous K run).
// ---------------------------------------------------------------------------
__global__ void __launch_bounds__(256)
prep_weights(const float* __restrict__ Wq, const float* __restrict__ Wk,
             const float* __restrict__ Wv, _Float16* __restrict__ wt) {
  int idx = blockIdx.x * 256 + threadIdx.x;          // 3*64*1024 total
  if (idx >= 3 * DH * DIN) return;
  int w   = idx >> 16;                               // / (64*1024)
  int rem = idx & 0xFFFF;
  int n   = rem >> 10;                               // output dim
  int k   = rem & 1023;                              // input dim
  const float* W = (w == 0) ? Wq : ((w == 1) ? Wk : Wv);
  wt[idx] = (_Float16)W[k * DH + n];
}

// ---------------------------------------------------------------------------
// Kernel 2: fused QKV projection.  One wave32 per 16-row tile of X.
//   qh[b][s][h] = f16( (X@Wq)*0.125 )   (softmax scale folded in)
//   kh[b][s][h] = f16( X@Wk )
//   vt[b][h][s] = f16( X@Wv )           (transposed for the P*V matmul)
// ---------------------------------------------------------------------------
__global__ void __launch_bounds__(32)
qkv_proj(const float* __restrict__ X, const _Float16* __restrict__ wt,
         _Float16* __restrict__ qh, _Float16* __restrict__ kh,
         _Float16* __restrict__ vt) {
  const int lane = threadIdx.x;
  const int tile = blockIdx.x;                       // 0 .. B*SEQ/16-1
  const int b    = tile >> 8;                        // SEQ/16 = 256 tiles/batch
  const int s0   = (tile & 255) << 4;
  const int nlo  = lane & 15;

  const float* xrow = X + ((size_t)(b * SEQ + s0 + nlo) * DIN);

  f8v acc[3][4];
#pragma unroll
  for (int w = 0; w < 3; ++w)
#pragma unroll
    for (int nt = 0; nt < 4; ++nt) acc[w][nt] = (f8v){};

  for (int kc = 0; kc < DIN; kc += 32) {
    h16v a = load_a_f32(xrow + kc, lane);
#pragma unroll
    for (int w = 0; w < 3; ++w) {
#pragma unroll
      for (int nt = 0; nt < 4; ++nt) {
        const _Float16* colp =
            wt + ((size_t)(w * DH + nt * 16 + nlo) * DIN) + kc;
        acc[w][nt] = wmma_f16(a, load_b_f16(colp, lane), acc[w][nt]);
      }
    }
  }

  // C/D layout: VGPR i -> row (lane<16 ? i : i+8), col = lane%16
  const int rbase = s0 + ((lane < 16) ? 0 : 8);
#pragma unroll
  for (int i = 0; i < 8; ++i) {
    const int row = rbase + i;
#pragma unroll
    for (int nt = 0; nt < 4; ++nt) {
      const int col = nt * 16 + nlo;
      qh[((size_t)b * SEQ + row) * DH + col] = (_Float16)(acc[0][nt][i] * 0.125f);
      kh[((size_t)b * SEQ + row) * DH + col] = (_Float16)acc[1][nt][i];
      vt[((size_t)b * DH + col) * SEQ + row] = (_Float16)acc[2][nt][i];
    }
  }
}

// ---------------------------------------------------------------------------
// Kernel 3: flash attention, one wave32 per 16-query tile.
//   Streams K/V (L2-resident f16) in 32-key chunks; online softmax in the
//   WMMA C-layout; P re-layout via LDS; P*V with WMMA.
// ---------------------------------------------------------------------------
__global__ void __launch_bounds__(32)
attn_fwd(const _Float16* __restrict__ qh, const _Float16* __restrict__ kh,
         const _Float16* __restrict__ vt, float* __restrict__ out) {
  __shared__ __align__(16) _Float16 pbuf[16 * 40];   // 16 rows, stride 40 halves

  const int lane = threadIdx.x;
  const int tile = blockIdx.x;
  const int b    = tile >> 8;
  const int s0   = (tile & 255) << 4;
  const int nlo  = lane & 15;
  const int rb   = (lane < 16) ? 0 : 8;              // C-layout row base

  const _Float16* Q = qh + (size_t)b * SEQ * DH;
  const _Float16* K = kh + (size_t)b * SEQ * DH;
  const _Float16* V = vt + (size_t)b * DH * SEQ;

  // Q tile A-operands for head-dim chunks [0,32) and [32,64)
  const _Float16* qrow = Q + (size_t)(s0 + nlo) * DH;
  h16v aq0 = load_a_f16(qrow, lane);
  h16v aq1 = load_a_f16(qrow + 32, lane);

  f8v o0 = (f8v){}, o1 = (f8v){}, o2 = (f8v){}, o3 = (f8v){};
  float m[8], l[8];
#pragma unroll
  for (int i = 0; i < 8; ++i) { m[i] = -3.0e38f; l[i] = 0.0f; }

  const int q_hi = s0 + 15;

  for (int kb = 0; kb <= q_hi; kb += 32) {
    // ---- scores S = (Q*1/8) K^T for keys [kb, kb+32) : 4 WMMAs ----
    f8v sc0 = (f8v){}, sc1 = (f8v){};
    {
      const _Float16* kcol0 = K + (size_t)(kb + nlo) * DH;
      const _Float16* kcol1 = K + (size_t)(kb + 16 + nlo) * DH;
      sc0 = wmma_f16(aq0, load_b_f16(kcol0, lane), sc0);
      sc0 = wmma_f16(aq1, load_b_f16(kcol0 + 32, lane), sc0);
      sc1 = wmma_f16(aq0, load_b_f16(kcol1, lane), sc1);
      sc1 = wmma_f16(aq1, load_b_f16(kcol1 + 32, lane), sc1);
      // hint next chunk toward the caches
      __builtin_prefetch(K + (size_t)(kb + 32 + nlo) * DH, 0, 0);
      __builtin_prefetch(V + (size_t)nlo * SEQ + kb + 32, 0, 0);
    }

    // ---- causal mask (only chunks overlapping the diagonal) ----
    if (kb + 31 > s0) {
#pragma unroll
      for (int i = 0; i < 8; ++i) {
        const int row = s0 + rb + i;
        if (kb + nlo > row)      sc0[i] = -3.0e38f;
        if (kb + 16 + nlo > row) sc1[i] = -3.0e38f;
      }
    }

    // ---- online softmax: per-row reductions across the 16-lane halves ----
#pragma unroll
    for (int i = 0; i < 8; ++i) {
      float rm = fmaxf(sc0[i], sc1[i]);
      rm = fmaxf(rm, __shfl_xor(rm, 1, 16));
      rm = fmaxf(rm, __shfl_xor(rm, 2, 16));
      rm = fmaxf(rm, __shfl_xor(rm, 4, 16));
      rm = fmaxf(rm, __shfl_xor(rm, 8, 16));
      const float mn = fmaxf(m[i], rm);
      const float c  = __expf(m[i] - mn);
      const float p0 = __expf(sc0[i] - mn);
      const float p1 = __expf(sc1[i] - mn);
      float rs = p0 + p1;
      rs += __shfl_xor(rs, 1, 16);
      rs += __shfl_xor(rs, 2, 16);
      rs += __shfl_xor(rs, 4, 16);
      rs += __shfl_xor(rs, 8, 16);
      l[i] = l[i] * c + rs;
      m[i] = mn;
      o0[i] *= c; o1[i] *= c; o2[i] *= c; o3[i] *= c;
      // spill P (C-layout) to LDS as f16, row-major stride 40
      const int row = rb + i;
      pbuf[row * 40 + nlo]      = (_Float16)p0;
      pbuf[row * 40 + 16 + nlo] = (_Float16)p1;
    }

    // make the cross-lane LDS traffic visible before the A-layout reload
    asm volatile("s_wait_dscnt 0" ::: "memory");

    // ---- reload P in WMMA A-layout (16x32) ----
    h16v ap;
    {
      const _Float16* pr = pbuf + nlo * 40;
      const int base = (lane < 16) ? 0 : 8;
      h8v lo = *(const h8v*)(pr + base);
      h8v hi = *(const h8v*)(pr + base + 16);
#pragma unroll
      for (int i = 0; i < 8; ++i) { ap[i] = lo[i]; ap[i + 8] = hi[i]; }
    }

    // ---- O += P * V : 4 WMMAs (V^T gives contiguous key-runs per column) ----
    o0 = wmma_f16(ap, load_b_f16(V + (size_t)(0 * 16 + nlo) * SEQ + kb, lane), o0);
    o1 = wmma_f16(ap, load_b_f16(V + (size_t)(1 * 16 + nlo) * SEQ + kb, lane), o1);
    o2 = wmma_f16(ap, load_b_f16(V + (size_t)(2 * 16 + nlo) * SEQ + kb, lane), o2);
    o3 = wmma_f16(ap, load_b_f16(V + (size_t)(3 * 16 + nlo) * SEQ + kb, lane), o3);
  }

  // ---- epilogue: out = O / l, f32 row-major [B,S,DH] ----
#pragma unroll
  for (int i = 0; i < 8; ++i) {
    const float inv = 1.0f / l[i];
    const size_t rowoff = ((size_t)b * SEQ + s0 + rb + i) * DH;
    out[rowoff + 0 * 16 + nlo] = o0[i] * inv;
    out[rowoff + 1 * 16 + nlo] = o1[i] * inv;
    out[rowoff + 2 * 16 + nlo] = o2[i] * inv;
    out[rowoff + 3 * 16 + nlo] = o3[i] * inv;
  }
}

// ---------------------------------------------------------------------------
// Launcher.  Workspace layout (bytes):
//   [0, 384K)          wt  : f16 [3][64][1024]
//   [384K, +2M)        qh  : f16 [B][S][64]   (pre-scaled by 1/8)
//   [+2M,  +2M)        kh  : f16 [B][S][64]
//   [+2M,  +2M)        vt  : f16 [B][64][S]
// Total ~6.4 MB.
// ---------------------------------------------------------------------------
extern "C" void kernel_launch(void* const* d_in, const int* in_sizes, int n_in,
                              void* d_out, int out_size, void* d_ws, size_t ws_size,
                              hipStream_t stream) {
  const float* X  = (const float*)d_in[0];
  const float* Wq = (const float*)d_in[1];
  const float* Wk = (const float*)d_in[2];
  const float* Wv = (const float*)d_in[3];
  float* out = (float*)d_out;

  char* ws = (char*)d_ws;
  const size_t WT_BYTES = (size_t)3 * DH * DIN * sizeof(_Float16);   // 384 KB
  const size_t QK_BYTES = (size_t)BATCH * SEQ * DH * sizeof(_Float16); // 2 MB
  _Float16* wt = (_Float16*)(ws);
  _Float16* qh = (_Float16*)(ws + WT_BYTES);
  _Float16* kh = (_Float16*)(ws + WT_BYTES + QK_BYTES);
  _Float16* vt = (_Float16*)(ws + WT_BYTES + 2 * QK_BYTES);

  prep_weights<<<(3 * DH * DIN) / 256, 256, 0, stream>>>(Wq, Wk, Wv, wt);
  qkv_proj<<<(BATCH * SEQ) / 16, 32, 0, stream>>>(X, wt, qh, kh, vt);
  attn_fwd<<<(BATCH * SEQ) / 16, 32, 0, stream>>>(qh, kh, vt, out);
}
